// Mamba3Block_76321568850050
// MI455X (gfx1250) — matrix operand order repaired and checked by reading, not compiled, not run on hardware
//
#include <hip/hip_runtime.h>
#include <hip/hip_bf16.h>

// ---------------- problem constants ----------------
#define B_  8
#define T_  4096
#define C_  512
#define DIN 1024           // D_INNER
#define HD  32             // HEADDIM
#define H_  32             // heads
#define N_  64             // state dim
#define DPJ 2208           // D_IN_PROJ = 2*1024 + 2*64 + 32
#define NTOK (B_ * T_)     // 32768
#define EPS 1e-5f

typedef __bf16 bf16_t;
typedef bf16_t v16bf __attribute__((ext_vector_type(16)));
typedef float  v8f   __attribute__((ext_vector_type(8)));

// ---------------- f32 -> bf16 convert ----------------
__global__ void cvt_bf16_kernel(const float* __restrict__ in,
                                __hip_bfloat16* __restrict__ out, int n) {
    int i = blockIdx.x * blockDim.x + threadIdx.x;
    if (i < n) out[i] = __float2bfloat16(in[i]);
}

// ---------------- RMSNorm (row=512) + cvt to bf16 ----------------
__global__ __launch_bounds__(256)
void rmsnorm_kernel(const float* __restrict__ x, const float* __restrict__ w,
                    __hip_bfloat16* __restrict__ out) {
    int row = blockIdx.x;                      // 0..NTOK-1
    int tid = threadIdx.x;                     // 256 threads, 2 elems each
    const float* xr = x + (size_t)row * C_;
    float2 v = ((const float2*)xr)[tid];
    __shared__ float red[256];
    red[tid] = v.x * v.x + v.y * v.y;
    __syncthreads();
    for (int s = 128; s > 0; s >>= 1) {
        if (tid < s) red[tid] += red[tid + s];
        __syncthreads();
    }
    float scale = rsqrtf(red[0] / (float)C_ + EPS);
    float2 wv = ((const float2*)w)[tid];
    size_t o = (size_t)row * C_ + 2 * tid;
    out[o]     = __float2bfloat16(v.x * scale * wv.x);
    out[o + 1] = __float2bfloat16(v.y * scale * wv.y);
}

// ---------------- WMMA bf16 GEMM: Cout[M,N] = A[M,K] * B[N,K]^T (+residual) ----
// Register-blocked: one wave -> 32x32 output (2 A-tiles x 2 B-tiles, 4 accs).
// Block = 256 threads = 8 waves stacked in M -> 256 x 32 block tile.
__global__ __launch_bounds__(256)
void gemm_bf16_kernel(const __hip_bfloat16* __restrict__ A,
                      const __hip_bfloat16* __restrict__ Bm,
                      const float* __restrict__ residual,
                      float* __restrict__ Cout,
                      int M, int N, int K) {
    const int wave = threadIdx.x >> 5;
    const int lane = threadIdx.x & 31;
    const int g    = lane >> 4;          // half-wave group
    const int l16  = lane & 15;
    const int m0 = (blockIdx.x * 8 + wave) * 32;
    const int n0 = blockIdx.y * 32;

    const bf16_t* Ab = (const bf16_t*)A;
    const bf16_t* Bb = (const bf16_t*)Bm;

    v8f acc00 = {}, acc01 = {}, acc10 = {}, acc11 = {};
    for (int k0 = 0; k0 < K; k0 += 32) {
        union { v16bf v; uint4 q[2]; } a0, a1, b0, b1;
        // A 16x32 bf16: lane(m=l16,g) holds K = [k0+8g, +8) and [k0+16+8g, +8)
        const bf16_t* ap0 = Ab + (size_t)(m0 + l16) * K + k0;
        const bf16_t* ap1 = ap0 + (size_t)16 * K;
        a0.q[0] = *(const uint4*)(ap0 + g * 8);
        a0.q[1] = *(const uint4*)(ap0 + 16 + g * 8);
        a1.q[0] = *(const uint4*)(ap1 + g * 8);
        a1.q[1] = *(const uint4*)(ap1 + 16 + g * 8);
        // B 32x16 bf16: lane(n=l16,g) holds K = [k0+8g, +8) and [k0+16+8g, +8)
        const bf16_t* bp0 = Bb + (size_t)(n0 + l16) * K + k0;
        const bf16_t* bp1 = bp0 + (size_t)16 * K;
        b0.q[0] = *(const uint4*)(bp0 + g * 8);
        b0.q[1] = *(const uint4*)(bp0 + 16 + g * 8);
        b1.q[0] = *(const uint4*)(bp1 + g * 8);
        b1.q[1] = *(const uint4*)(bp1 + 16 + g * 8);
        acc00 = __builtin_amdgcn_wmma_f32_16x16x32_bf16(
            false, a0.v, false, b0.v, (short)0, acc00, false, false);
        acc01 = __builtin_amdgcn_wmma_f32_16x16x32_bf16(
            false, a0.v, false, b1.v, (short)0, acc01, false, false);
        acc10 = __builtin_amdgcn_wmma_f32_16x16x32_bf16(
            false, a1.v, false, b0.v, (short)0, acc10, false, false);
        acc11 = __builtin_amdgcn_wmma_f32_16x16x32_bf16(
            false, a1.v, false, b1.v, (short)0, acc11, false, false);
    }
    // D layout per tile: VGPR r -> row base + r + 8g, col base + l16
    #pragma unroll
    for (int mt = 0; mt < 2; mt++) {
        #pragma unroll
        for (int nt = 0; nt < 2; nt++) {
            v8f acc = (mt == 0) ? (nt == 0 ? acc00 : acc01)
                                : (nt == 0 ? acc10 : acc11);
            const int mb = m0 + mt * 16 + g * 8;
            const int nc = n0 + nt * 16 + l16;
            #pragma unroll
            for (int r = 0; r < 8; r++) {
                size_t idx = (size_t)(mb + r) * N + nc;
                float v = acc[r];
                if (residual) v += residual[idx];
                Cout[idx] = v;
            }
        }
    }
}

// ---------------- per-token scan prep: dt, a, RMS(B), RMS(C) ----------------
__global__ __launch_bounds__(64)
void prep_scan_kernel(const float* __restrict__ zx,
                      const float* __restrict__ dt_bias,
                      const float* __restrict__ A_log,
                      const float* __restrict__ Bw,
                      const float* __restrict__ Cw,
                      float* __restrict__ dtA, float* __restrict__ aA,
                      float* __restrict__ Bn, float* __restrict__ Cn) {
    int tok = blockIdx.x;
    int tid = threadIdx.x;                       // 64 threads
    const float* row = zx + (size_t)tok * DPJ;
    if (tid < H_) {
        float raw = row[2 * DIN + 2 * N_ + tid] + dt_bias[tid];
        float dt  = (raw > 20.f) ? raw : log1pf(expf(raw));   // softplus
        float a   = expf(-expf(A_log[tid]) * dt);
        dtA[(size_t)tok * H_ + tid] = dt;
        aA [(size_t)tok * H_ + tid] = a;
    }
    float bv = row[2 * DIN + tid];
    float cv = row[2 * DIN + N_ + tid];
    __shared__ float rb[64], rc[64];
    rb[tid] = bv * bv; rc[tid] = cv * cv;
    __syncthreads();
    for (int s = 32; s > 0; s >>= 1) {
        if (tid < s) { rb[tid] += rb[tid + s]; rc[tid] += rc[tid + s]; }
        __syncthreads();
    }
    float sb = rsqrtf(rb[0] / (float)N_ + EPS);
    float sc = rsqrtf(rc[0] / (float)N_ + EPS);
    Bn[(size_t)tok * N_ + tid] = bv * sb * Bw[tid];
    Cn[(size_t)tok * N_ + tid] = cv * sc * Cw[tid];
}

// ---------------- sequential scan: one wave32 per (b,h) ----------------
// lane = head-dim p; full N=64 state per lane in registers (unrolled).
// Software-pipelined: step t+1's global data is prefetched into registers
// while step t computes from double-buffered LDS (one barrier per step).
__global__ __launch_bounds__(32)
void scan_kernel(const float* __restrict__ zx,
                 const float* __restrict__ dtA, const float* __restrict__ aA,
                 const float* __restrict__ Bn,  const float* __restrict__ Cn,
                 const float* __restrict__ omega, const float* __restrict__ Dv,
                 __hip_bfloat16* __restrict__ ybf) {
    const int bh = blockIdx.x;
    const int b  = bh >> 5;
    const int h  = bh & 31;
    const int lane = threadIdx.x;                // p = 0..31

    float hs[N_];
    #pragma unroll
    for (int i = 0; i < N_; i++) hs[i] = 0.f;

    const float om = omega[h * (N_ / 2) + lane]; // lane owns rotation pair 'lane'
    const float Dh = Dv[h];
    const float* zxb = zx + (size_t)b * T_ * DPJ;
    const size_t tok0 = (size_t)b * T_;

    __shared__ float sCS[2][N_];   // interleaved cos/sin per pair, double-buffered
    __shared__ float sB[2][N_];
    __shared__ float sC[2][N_];

    // preload step 0
    float  dt_c = dtA[tok0 * H_ + h];
    float  at_c = aA [tok0 * H_ + h];
    float2 bv_c = ((const float2*)(Bn + tok0 * N_))[lane];
    float2 cv_c = ((const float2*)(Cn + tok0 * N_))[lane];
    float  xv_c = zxb[(size_t)0 * DPJ + DIN + h * HD + lane];
    float  zv_c = zxb[(size_t)0 * DPJ +        h * HD + lane];

    for (int t = 0; t < T_; t++) {
        const int buf = t & 1;
        // rotation coefficients for this step (lane owns pair 'lane')
        float th = dt_c * om;
        float sn = __sinf(th);
        float cn = __cosf(th);
        ((float2*)sCS[buf])[lane] = make_float2(cn, sn);
        ((float2*)sB[buf])[lane]  = bv_c;
        ((float2*)sC[buf])[lane]  = cv_c;
        __syncthreads();

        // prefetch step t+1 (clamped; overlaps with the compute below)
        const int tn = (t + 1 < T_) ? (t + 1) : (T_ - 1);
        const size_t tokn = tok0 + tn;
        float  dt_n = dtA[tokn * H_ + h];
        float  at_n = aA [tokn * H_ + h];
        float2 bv_n = ((const float2*)(Bn + tokn * N_))[lane];
        float2 cv_n = ((const float2*)(Cn + tokn * N_))[lane];
        float  xv_n = zxb[(size_t)tn * DPJ + DIN + h * HD + lane];
        float  zv_n = zxb[(size_t)tn * DPJ +        h * HD + lane];

        const float dtx = dt_c * xv_c;
        const float at  = at_c;
        float y = 0.f;
        #pragma unroll
        for (int n = 0; n < N_ / 2; n++) {
            float c2 = sCS[buf][2 * n], s2 = sCS[buf][2 * n + 1];
            float h0 = hs[2 * n], h1 = hs[2 * n + 1];
            float r0 = h0 * c2 - h1 * s2;
            float r1 = h0 * s2 + h1 * c2;
            float u0 = at * r0 + dtx * sB[buf][2 * n];
            float u1 = at * r1 + dtx * sB[buf][2 * n + 1];
            hs[2 * n] = u0; hs[2 * n + 1] = u1;
            y += u0 * sC[buf][2 * n] + u1 * sC[buf][2 * n + 1];
        }

        float yg = (y + Dh * xv_c) * zv_c;       // + D*x, gate with z
        ybf[(tok0 + t) * DIN + h * HD + lane] = __float2bfloat16(yg);

        // rotate pipeline registers
        dt_c = dt_n; at_c = at_n; bv_c = bv_n; cv_c = cv_n;
        xv_c = xv_n; zv_c = zv_n;
    }
}

// ---------------- host side ----------------
extern "C" void kernel_launch(void* const* d_in, const int* in_sizes, int n_in,
                              void* d_out, int out_size, void* d_ws, size_t ws_size,
                              hipStream_t stream) {
    const float* input     = (const float*)d_in[0];
    const float* rms_w     = (const float*)d_in[1];
    const float* in_proj_W = (const float*)d_in[2];   // (DPJ, C)
    const float* dt_bias   = (const float*)d_in[3];
    const float* A_log     = (const float*)d_in[4];
    const float* omega     = (const float*)d_in[5];   // (H, N/2)
    const float* Dvec      = (const float*)d_in[6];
    const float* Bnorm_w   = (const float*)d_in[7];
    const float* Cnorm_w   = (const float*)d_in[8];
    const float* out_proj_W= (const float*)d_in[9];   // (C, DIN)
    float* out = (float*)d_out;

    char* ws = (char*)d_ws;
    size_t off = 0;
    auto take = [&](size_t bytes) -> void* {
        void* p = ws + off;
        off = (off + bytes + 255) & ~(size_t)255;
        return p;
    };
    __hip_bfloat16* normed_bf = (__hip_bfloat16*)take((size_t)NTOK * C_ * 2);
    __hip_bfloat16* Wa_bf     = (__hip_bfloat16*)take((size_t)DPJ * C_ * 2);
    __hip_bfloat16* Wb_bf     = (__hip_bfloat16*)take((size_t)C_ * DIN * 2);
    float*          zxbcdt    = (float*)take((size_t)NTOK * DPJ * 4);
    float*          dtA       = (float*)take((size_t)NTOK * H_ * 4);
    float*          aA        = (float*)take((size_t)NTOK * H_ * 4);
    float*          Bn        = (float*)take((size_t)NTOK * N_ * 4);
    float*          Cn        = (float*)take((size_t)NTOK * N_ * 4);
    __hip_bfloat16* ybf       = (__hip_bfloat16*)take((size_t)NTOK * DIN * 2);
    (void)ws_size; (void)n_in; (void)in_sizes; (void)out_size;

    // 1) weights -> bf16
    {
        int n1 = DPJ * C_;
        cvt_bf16_kernel<<<(n1 + 255) / 256, 256, 0, stream>>>(in_proj_W, Wa_bf, n1);
        int n2 = C_ * DIN;
        cvt_bf16_kernel<<<(n2 + 255) / 256, 256, 0, stream>>>(out_proj_W, Wb_bf, n2);
    }
    // 2) RMSNorm(input) -> bf16
    rmsnorm_kernel<<<NTOK, 256, 0, stream>>>(input, rms_w, normed_bf);
    // 3) in_proj GEMM: (NTOK x 512) x (2208 x 512)^T -> zxbcdt f32
    {
        dim3 grid(NTOK / 256, DPJ / 32);   // (128, 69)
        gemm_bf16_kernel<<<grid, 256, 0, stream>>>(normed_bf, Wa_bf, nullptr,
                                                   zxbcdt, NTOK, DPJ, C_);
    }
    // 4) per-token scan prep
    prep_scan_kernel<<<NTOK, 64, 0, stream>>>(zxbcdt, dt_bias, A_log,
                                              Bnorm_w, Cnorm_w, dtA, aA, Bn, Cn);
    // 5) sequential scan, gated bf16 output
    scan_kernel<<<B_ * H_, 32, 0, stream>>>(zxbcdt, dtA, aA, Bn, Cn,
                                            omega, Dvec, ybf);
    // 6) out_proj GEMM + residual: (NTOK x 1024) x (512 x 1024)^T + input -> out
    {
        dim3 grid(NTOK / 256, C_ / 32);    // (128, 16)
        gemm_bf16_kernel<<<grid, 256, 0, stream>>>(ybf, Wb_bf, input,
                                                   out, NTOK, C_, DIN);
    }
}
